// EdgeWeightedMultiHeadSelfAttention_5944234738057
// MI455X (gfx1250) — matrix operand order repaired and checked by reading, not compiled, not run on hardware
//
#include <hip/hip_runtime.h>
#include <hip/hip_bf16.h>

// EdgeWeightedMultiHeadSelfAttention for MI455X (gfx1250), B=2 N=512 E=128 H=8 D=16.
// Stage 1: qkv = x @ w_qkv^T          -> WMMA bf16 GEMM  [1024x128]x[128x384]
// Stage 2: fused edge attention       -> VALU + LDS (ekv NEVER materialized)
// Stage 3: out = hat @ w_prj^T + b    -> WMMA bf16 GEMM  [1024x128]x[128x128]

#define E_DIM 128
#define NSEQ  512
#define NHEAD 8
#define DHEAD 16
#define HD    128
#define HD3   384

typedef __attribute__((ext_vector_type(16))) __bf16 v16bf;
typedef __attribute__((ext_vector_type(8)))  float  v8f;

union ABFrag {
    v16bf v;
    unsigned short u[16];
};

// fp32 -> bf16, round-to-nearest-even
__device__ __forceinline__ unsigned short f2bf(float f) {
    unsigned int u = __float_as_uint(f);
    u += 0x7FFFu + ((u >> 16) & 1u);
    return (unsigned short)(u >> 16);
}

// Out[m,c] = sum_k A[m,k] * W[c,k] (+ bias[c]).  A:[M,K] W:[C,K] row-major.
// One wave (32 lanes) per 16x16 output tile, v_wmma_f32_16x16x32_bf16 over K.
// K and HAS_BIAS are compile-time: the K-loop (K/32 WMMAs) fully unrolls so the
// scheduler overlaps next-step loads/bf16-packs with the current WMMA, and the
// bias path carries no per-tile pointer-null branch.
// Fragment layouts per CDNA5 ISA 7.12.2 (wave32):
//   A (16x32 bf16): lanes 0-15 = rows; VGPR j<4 -> K=2j (+8 hi half), j>=4 -> K=16+2(j-4) (+8 hi half)
//   B (32x16 bf16): lanes 0-15 = cols; VGPR j<4 -> K=2j (+8 hi half), j>=4 -> K=16+2(j-4) (+8 hi half)
//   C/D (16x16 f32): VGPR r -> M = r + 8*half, N = lane%16
template <int K, bool HAS_BIAS>
__global__ __launch_bounds__(256) void wmma_gemm_nt(
    const float* __restrict__ A, const float* __restrict__ W,
    const float* __restrict__ bias, float* __restrict__ Out,
    int M, int C)
{
    const int wave = threadIdx.x >> 5;
    const int lane = threadIdx.x & 31;
    const int gw   = blockIdx.x * 8 + wave;
    const int Ct   = C >> 4;
    const int mt   = gw / Ct;
    const int ct   = gw % Ct;
    if (mt * 16 >= M) return;                 // wave-uniform: EXEC stays all-ones for WMMA

    const int half = lane >> 4;
    const int row  = lane & 15;

    const float* arow = A + (size_t)(mt * 16 + row) * K;   // A row = output row M
    const float* wrow = W + (size_t)(ct * 16 + row) * K;   // W row = output col C

    v8f acc = {};
#pragma unroll
    for (int k0 = 0; k0 < K; k0 += 32) {
        ABFrag a, b;
#pragma unroll
        for (int j = 0; j < 8; ++j) {
            const int ka = k0 + ((j < 4) ? 2 * j : 16 + 2 * (j - 4)) + half * 8;
            const float2 fa = *(const float2*)(arow + ka);
            a.u[2 * j]     = f2bf(fa.x);
            a.u[2 * j + 1] = f2bf(fa.y);
            const int kb = k0 + ((j < 4) ? 2 * j : 16 + 2 * (j - 4)) + half * 8;
            const float2 fb = *(const float2*)(wrow + kb);
            b.u[2 * j]     = f2bf(fb.x);
            b.u[2 * j + 1] = f2bf(fb.y);
        }
        acc = __builtin_amdgcn_wmma_f32_16x16x32_bf16(
            /*neg_a=*/false, a.v, /*neg_b=*/false, b.v,
            /*c_mod=*/(short)0, acc, /*reuse_a=*/false, /*reuse_b=*/false);
    }

    const int c = ct * 16 + row;
    float bb = 0.0f;
    if (HAS_BIAS) bb = bias[c];
    // Single base pointer; per-r offsets are compile-time multiples of C.
    float* obase = Out + (size_t)(mt * 16 + half * 8) * C + c;
#pragma unroll
    for (int r = 0; r < 8; ++r) {
        obase[(size_t)r * C] = acc[r] + bb;
    }
}

// Fused edge attention: one 256-thread block per (b,n).
//   dots[h,m] = q[h]·k[h] + sum_d q[h,d]*leaky(e[m]*wk[h,d]+bk[h,d]);  softmax over m
//   hat[h,d]  = v[h,d] + (1/Σexp) * sum_m exp[h,m]*leaky(e[m]*wv[h,d]+bv[h,d])
__global__ __launch_bounds__(256) void edge_attn(
    const float* __restrict__ qkv,   // [B*N, 384] = q|k|v
    const float* __restrict__ e,     // [B*N, N]
    const float* __restrict__ w_ekv, // [256] (wk|wv)
    const float* __restrict__ b_ekv, // [256] (bk|bv)
    float* __restrict__ hat)         // [B*N, 128]
{
    const int bn  = blockIdx.x;
    const int tid = threadIdx.x;

    __shared__ float s_q[HD], s_v[HD];
    __shared__ float s_wk[HD], s_bk[HD], s_wv[HD], s_bv[HD];
    __shared__ float s_prod[HD];
    __shared__ float s_qk[NHEAD];
    __shared__ float s_inv[NHEAD];
    __shared__ float s_e[NSEQ];
    __shared__ float s_dots[NHEAD][NSEQ];   // 16 KB
    __shared__ float s_part[2][HD];

    const float* qrow = qkv + (size_t)bn * HD3;
    const float* erow = e   + (size_t)bn * NSEQ;

    // prefetch the whole e-row (gfx1250 global_prefetch_b8)
    if (tid < 32) __builtin_prefetch(erow + tid * 16, 0, 0);

    if (tid < HD) {
        const float qv = qrow[tid];
        const float kv = qrow[HD + tid];
        s_q[tid]    = qv;
        s_v[tid]    = qrow[2 * HD + tid];
        s_prod[tid] = qv * kv;
        s_wk[tid]   = w_ekv[tid];
        s_bk[tid]   = b_ekv[tid];
        s_wv[tid]   = w_ekv[HD + tid];
        s_bv[tid]   = b_ekv[HD + tid];
    }
    s_e[tid]       = erow[tid];
    s_e[tid + 256] = erow[tid + 256];
    __syncthreads();

    if (tid < NHEAD) {
        float s = 0.0f;
#pragma unroll
        for (int d = 0; d < DHEAD; ++d) s += s_prod[tid * DHEAD + d];
        s_qk[tid] = s;
    }
    __syncthreads();

    // dots: each thread handles 2 m-values x 8 heads x 16 d
#pragma unroll
    for (int mi = 0; mi < 2; ++mi) {
        const int m = tid + mi * 256;
        const float ev = s_e[m];
#pragma unroll
        for (int h = 0; h < NHEAD; ++h) {
            float acc = s_qk[h];
#pragma unroll
            for (int d = 0; d < DHEAD; ++d) {
                const int i = h * DHEAD + d;
                float z = fmaf(ev, s_wk[i], s_bk[i]);
                z = (z > 0.0f) ? z : 0.01f * z;         // leaky_relu(0.01)
                acc = fmaf(s_q[i], z, acc);
            }
            s_dots[h][m] = acc * 0.25f;                  // 1/sqrt(D)
        }
    }
    __syncthreads();

    // softmax over m: one wave32 per head
    {
        const int h    = tid >> 5;
        const int lane = tid & 31;
        float mx = -1e30f;
        for (int m = lane; m < NSEQ; m += 32) mx = fmaxf(mx, s_dots[h][m]);
#pragma unroll
        for (int off = 16; off > 0; off >>= 1) mx = fmaxf(mx, __shfl_xor(mx, off, 32));
        float sum = 0.0f;
        for (int m = lane; m < NSEQ; m += 32) {
            const float p = __expf(s_dots[h][m] - mx);
            s_dots[h][m] = p;                            // keep unnormalized exp
            sum += p;
        }
#pragma unroll
        for (int off = 16; off > 0; off >>= 1) sum += __shfl_xor(sum, off, 32);
        if (lane == 0) s_inv[h] = 1.0f / sum;
    }
    __syncthreads();

    // hat accumulation: 2 threads per (h,d), each over half the m range
    {
        const int i  = tid & 127;        // h*16+d
        const int mh = tid >> 7;         // which m-half
        const int h  = i >> 4;
        const float wv = s_wv[i], bv = s_bv[i];
        float acc = 0.0f;
        const int m0 = mh * 256;
        for (int m = m0; m < m0 + 256; ++m) {
            float z = fmaf(s_e[m], wv, bv);
            z = (z > 0.0f) ? z : 0.01f * z;
            acc = fmaf(s_dots[h][m], z, acc);
        }
        s_part[mh][i] = acc;
    }
    __syncthreads();

    if (tid < HD) {
        const int h = tid >> 4;
        hat[(size_t)bn * HD + tid] =
            s_v[tid] + (s_part[0][tid] + s_part[1][tid]) * s_inv[h];
    }
}

extern "C" void kernel_launch(void* const* d_in, const int* in_sizes, int n_in,
                              void* d_out, int out_size, void* d_ws, size_t ws_size,
                              hipStream_t stream) {
    const float* x     = (const float*)d_in[0];
    const float* e     = (const float*)d_in[1];
    const float* w_qkv = (const float*)d_in[2];
    const float* w_ekv = (const float*)d_in[3];
    const float* b_ekv = (const float*)d_in[4];
    const float* w_prj = (const float*)d_in[5];
    const float* b_prj = (const float*)d_in[6];
    float* out = (float*)d_out;

    const int BN = in_sizes[0] / E_DIM;          // B*N = 1024

    float* qkv = (float*)d_ws;                   // [BN, 384]  (1.5 MB)
    float* hat = qkv + (size_t)BN * HD3;         // [BN, 128]  (0.5 MB)

    // Stage 1: qkv projection  (tiles = 64*24 = 1536 waves, 8 waves/block)
    {
        const int tiles = (BN / 16) * (HD3 / 16);
        wmma_gemm_nt<E_DIM, false><<<(tiles + 7) / 8, 256, 0, stream>>>(
            x, w_qkv, nullptr, qkv, BN, HD3);
    }
    // Stage 2: fused edge attention, one block per (b,n)
    edge_attn<<<BN, 256, 0, stream>>>(qkv, e, w_ekv, b_ekv, hat);
    // Stage 3: output projection + bias (tiles = 64*8 = 512 waves)
    {
        const int tiles = (BN / 16) * (E_DIM / 16);
        wmma_gemm_nt<E_DIM, true><<<(tiles + 7) / 8, 256, 0, stream>>>(
            hat, w_prj, b_prj, out, BN, E_DIM);
    }
}
